// Length_Regulator_89421219103447
// MI455X (gfx1250) — compile-verified, matrix operand order checked by reading, not run on hardware
//
#include <hip/hip_runtime.h>

typedef __attribute__((ext_vector_type(16))) __bf16 v16bf;
typedef __attribute__((ext_vector_type(8)))  __bf16 v8bf;
typedef __attribute__((ext_vector_type(8)))  float  v8f;

#define B_   16
#define L_   512
#define D_   384
#define CD_  1536
#define LPAD 514          // 512 + one zero row each side
#define M_   (B_ * L_)    // 8192 output positions
#define MEL  4096

// ---------------------------------------------------------------------------
// Pack conv weights (O, I, 3) fp32 -> bf16 B-fragment layout:
//   wf[((tap*KT + kt)*NT + nt)*512 + lane*16 + j]
// lane = {n_lane(0..15), khalf}, K(j,khalf) = kt*32 + (j<8 ? j : j+8) + khalf*8
// ---------------------------------------------------------------------------
template<int K, int N>
__global__ void pack_w_kernel(const float* __restrict__ w, __bf16* __restrict__ wf) {
    constexpr int KT = K / 32, NT = N / 16;
    int id = blockIdx.x * blockDim.x + threadIdx.x;
    if (id >= 3 * KT * NT * 512) return;
    int j    = id & 15;
    int lane = (id >> 4) & 31;
    int rest = id >> 9;
    int nt   = rest % NT; rest /= NT;
    int kt   = rest % KT;
    int tap  = rest / KT;
    int khalf = lane >> 4;
    int kk = kt * 32 + ((j < 8) ? j : j + 8) + khalf * 8;
    int o  = nt * 16 + (lane & 15);
    wf[id] = (__bf16)w[((size_t)o * K + kk) * 3 + tap];
}

// fp32 (B,L,D) -> bf16 padded (B,LPAD,D) with zero rows at slots 0 and 513
__global__ void pad_x_kernel(const float* __restrict__ x, __bf16* __restrict__ xpad) {
    int id = blockIdx.x * blockDim.x + threadIdx.x;
    if (id >= B_ * LPAD * D_) return;
    int d = id % D_;
    int p = (id / D_) % LPAD;
    int b = id / (D_ * LPAD);
    float v = 0.f;
    if (p >= 1 && p <= L_) v = x[((size_t)b * L_ + (p - 1)) * D_ + d];
    xpad[id] = (__bf16)v;
}

__global__ void zero_edges_kernel(__bf16* __restrict__ pad, int N) {
    int id = blockIdx.x * blockDim.x + threadIdx.x;
    if (id >= B_ * 2 * N) return;
    int n  = id % N;
    int wb = id / N;
    int b  = wb >> 1;
    int slot = (wb & 1) ? (LPAD - 1) : 0;
    pad[(size_t)(b * LPAD + slot) * N + n] = (__bf16)0.f;
}

// ---------------------------------------------------------------------------
// Conv1d(k=3, SAME) as GEMM over padded activations. Each wave: 32x32 tile.
// Apad: bf16 [B*LPAD, K]; Wfrag: packed fragments; out is relu(conv + bias),
// either bf16 into a padded buffer (feeds next conv) or fp32 flat (feeds LN).
// ---------------------------------------------------------------------------
template<int K, int N, bool OUT_PAD>
__global__ __launch_bounds__(256) void gemm_conv3_kernel(
    const __bf16* __restrict__ Apad,
    const __bf16* __restrict__ Wfrag,
    const float*  __restrict__ bias,
    __bf16* __restrict__ OutPad,
    float*  __restrict__ OutF32)
{
    constexpr int KT = K / 32;
    constexpr int NT = N / 16;
    constexpr int WN = N / 32;                 // 32-wide n units
    const int wave = (int)((blockIdx.x * (size_t)blockDim.x + threadIdx.x) >> 5);
    const int lane = threadIdx.x & 31;
    const int mt2 = wave / WN;
    const int nt2 = wave % WN;
    const int m0 = mt2 * 32;
    const int n0 = nt2 * 32;
    const int rowa  = lane & 15;
    const int khalf = lane >> 4;

    v8f acc00 = {}, acc01 = {}, acc10 = {}, acc11 = {};

    const int m0a = m0 + rowa;
    const int m1a = m0 + 16 + rowa;
    const size_t prow0 = (size_t)((m0a >> 9) * LPAD + (m0a & 511));
    const size_t prow1 = (size_t)((m1a >> 9) * LPAD + (m1a & 511));

    for (int tap = 0; tap < 3; ++tap) {
        const __bf16* arow0 = Apad + (prow0 + tap) * K + khalf * 8;
        const __bf16* arow1 = Apad + (prow1 + tap) * K + khalf * 8;
        const __bf16* wbase = Wfrag + ((size_t)tap * KT) * NT * 512;
        #pragma unroll 2
        for (int kt = 0; kt < KT; ++kt) {
            union { v16bf v; v8bf h[2]; } a0, a1;
            a0.h[0] = *(const v8bf*)(arow0 + kt * 32);
            a0.h[1] = *(const v8bf*)(arow0 + kt * 32 + 16);
            a1.h[0] = *(const v8bf*)(arow1 + kt * 32);
            a1.h[1] = *(const v8bf*)(arow1 + kt * 32 + 16);
            const __bf16* wp = wbase + ((size_t)kt * NT + nt2 * 2) * 512 + lane * 16;
            v16bf b0 = *(const v16bf*)(wp);
            v16bf b1 = *(const v16bf*)(wp + 512);
            acc00 = __builtin_amdgcn_wmma_f32_16x16x32_bf16(false, a0.v, false, b0, (short)0, acc00, false, false);
            acc01 = __builtin_amdgcn_wmma_f32_16x16x32_bf16(false, a0.v, false, b1, (short)0, acc01, false, false);
            acc10 = __builtin_amdgcn_wmma_f32_16x16x32_bf16(false, a1.v, false, b0, (short)0, acc10, false, false);
            acc11 = __builtin_amdgcn_wmma_f32_16x16x32_bf16(false, a1.v, false, b1, (short)0, acc11, false, false);
        }
    }

    // C/D layout: lanes 0-15 -> N=lane, rows M = vgpr r; lanes 16-31 -> rows r+8
    const int nl = lane & 15;
    const int mh = (lane >> 4) << 3;
    auto epil = [&](v8f c, int mi, int nj) {
        const int n = n0 + nj * 16 + nl;
        const float bv = bias[n];
        const int mbase = m0 + mi * 16 + mh;
        #pragma unroll
        for (int r = 0; r < 8; ++r) {
            float v = c[r] + bv;
            v = v > 0.f ? v : 0.f;                 // ReLU
            const int m = mbase + r;
            if constexpr (OUT_PAD) {
                OutPad[(size_t)((m >> 9) * LPAD + (m & 511) + 1) * N + n] = (__bf16)v;
            } else {
                OutF32[(size_t)m * N + n] = v;
            }
        }
    };
    epil(acc00, 0, 0); epil(acc01, 0, 1); epil(acc10, 1, 0); epil(acc11, 1, 1);
}

// ---------------------------------------------------------------------------
// LayerNorm over D_=384, one wave per row (12 elems/lane)
// ---------------------------------------------------------------------------
__global__ __launch_bounds__(256) void ln1_kernel(
    const float* __restrict__ X, const float* __restrict__ g,
    const float* __restrict__ be, __bf16* __restrict__ Opad)
{
    int wid  = (int)((blockIdx.x * (size_t)blockDim.x + threadIdx.x) >> 5);
    int lane = threadIdx.x & 31;
    if (wid >= M_) return;
    const float* row = X + (size_t)wid * D_;
    float vals[12], s1 = 0.f, s2 = 0.f;
    #pragma unroll
    for (int k = 0; k < 12; ++k) { float v = row[lane + k * 32]; vals[k] = v; s1 += v; s2 += v * v; }
    #pragma unroll
    for (int o = 16; o > 0; o >>= 1) { s1 += __shfl_xor(s1, o, 32); s2 += __shfl_xor(s2, o, 32); }
    float mean = s1 * (1.f / D_);
    float rstd = rsqrtf(s2 * (1.f / D_) - mean * mean + 1e-5f);
    int b = wid >> 9, l = wid & 511;
    __bf16* orow = Opad + (size_t)(b * LPAD + l + 1) * D_;
    #pragma unroll
    for (int k = 0; k < 12; ++k) {
        int d = lane + k * 32;
        orow[d] = (__bf16)((vals[k] - mean) * rstd * g[d] + be[d]);
    }
}

// LayerNorm + fused (h @ wl + bl) projection -> dur_preds
__global__ __launch_bounds__(256) void ln2_linear_kernel(
    const float* __restrict__ X, const float* __restrict__ g,
    const float* __restrict__ be, const float* __restrict__ wl,
    const float* __restrict__ bl, float* __restrict__ dur)
{
    int wid  = (int)((blockIdx.x * (size_t)blockDim.x + threadIdx.x) >> 5);
    int lane = threadIdx.x & 31;
    if (wid >= M_) return;
    const float* row = X + (size_t)wid * D_;
    float vals[12], s1 = 0.f, s2 = 0.f;
    #pragma unroll
    for (int k = 0; k < 12; ++k) { float v = row[lane + k * 32]; vals[k] = v; s1 += v; s2 += v * v; }
    #pragma unroll
    for (int o = 16; o > 0; o >>= 1) { s1 += __shfl_xor(s1, o, 32); s2 += __shfl_xor(s2, o, 32); }
    float mean = s1 * (1.f / D_);
    float rstd = rsqrtf(s2 * (1.f / D_) - mean * mean + 1e-5f);
    float dot = 0.f;
    #pragma unroll
    for (int k = 0; k < 12; ++k) {
        int d = lane + k * 32;
        dot += ((vals[k] - mean) * rstd * g[d] + be[d]) * wl[d];
    }
    #pragma unroll
    for (int o = 16; o > 0; o >>= 1) dot += __shfl_xor(dot, o, 32);
    if (lane == 0) dur[wid] = dot + bl[0];
}

// ---------------------------------------------------------------------------
// Length regulator: cumsum, searchsorted, masked gather
// ---------------------------------------------------------------------------
__global__ void cumsum_kernel(const int* __restrict__ tgt, int* __restrict__ cum) {
    int b = threadIdx.x;
    if (b >= B_) return;
    int run = 0;
    for (int l = 0; l < L_; ++l) { run += tgt[b * L_ + l]; cum[b * L_ + l] = run; }
}

__global__ void searchsorted_kernel(const int* __restrict__ cum, int* __restrict__ idx) {
    int id = blockIdx.x * blockDim.x + threadIdx.x;
    if (id >= B_ * MEL) return;
    int b = id / MEL, t = id % MEL;
    const int* c = cum + b * L_;
    int lo = 0, hi = L_;
    while (lo < hi) { int mid = (lo + hi) >> 1; if (c[mid] <= t) lo = mid + 1; else hi = mid; }
    idx[id] = lo < (L_ - 1) ? lo : (L_ - 1);
}

__global__ void gather_kernel(const float* __restrict__ x, const int* __restrict__ idx,
                              const int* __restrict__ cum, float* __restrict__ out) {
    constexpr int Q = D_ / 4;  // float4 chunks per row
    int id = blockIdx.x * blockDim.x + threadIdx.x;
    if (id >= B_ * MEL * Q) return;
    int q = id % Q;
    int t = (id / Q) % MEL;
    int b = id / (Q * MEL);
    int total = cum[b * L_ + L_ - 1];
    float4 v = make_float4(0.f, 0.f, 0.f, 0.f);
    if (t < total) {
        int r = idx[b * MEL + t];
        v = *(const float4*)(x + ((size_t)b * L_ + r) * D_ + q * 4);
    }
    *(float4*)(out + ((size_t)b * MEL + t) * D_ + q * 4) = v;
}

// ---------------------------------------------------------------------------
extern "C" void kernel_launch(void* const* d_in, const int* in_sizes, int n_in,
                              void* d_out, int out_size, void* d_ws, size_t ws_size,
                              hipStream_t stream) {
    (void)in_sizes; (void)n_in; (void)out_size; (void)ws_size;
    const float* x      = (const float*)d_in[0];
    const int*   target = (const int*)  d_in[1];
    const float* w1  = (const float*)d_in[3];
    const float* b1  = (const float*)d_in[4];
    const float* w2  = (const float*)d_in[5];
    const float* b2  = (const float*)d_in[6];
    const float* g1  = (const float*)d_in[7];
    const float* be1 = (const float*)d_in[8];
    const float* w3  = (const float*)d_in[9];
    const float* b3  = (const float*)d_in[10];
    const float* w4  = (const float*)d_in[11];
    const float* b4  = (const float*)d_in[12];
    const float* g2  = (const float*)d_in[13];
    const float* be2 = (const float*)d_in[14];
    const float* wl  = (const float*)d_in[15];
    const float* bl  = (const float*)d_in[16];

    float* out = (float*)d_out;                        // (B, MEL, D) fp32
    float* dur = out + (size_t)B_ * MEL * D_;          // (B, L) fp32

    char* ws = (char*)d_ws;
    size_t off = 0;
    auto alloc = [&](size_t bytes) -> char* {
        char* p = ws + off;
        off = (off + bytes + 255) & ~(size_t)255;
        return p;
    };
    __bf16* wf1   = (__bf16*)alloc((size_t)3 * D_  * CD_ * 2);
    __bf16* wf2   = (__bf16*)alloc((size_t)3 * CD_ * D_  * 2);
    __bf16* wf3   = (__bf16*)alloc((size_t)3 * D_  * CD_ * 2);
    __bf16* wf4   = (__bf16*)alloc((size_t)3 * CD_ * D_  * 2);
    __bf16* xpad  = (__bf16*)alloc((size_t)B_ * LPAD * D_  * 2);  // reused as h1pad
    __bf16* y1pad = (__bf16*)alloc((size_t)B_ * LPAD * CD_ * 2);  // FFN mid, both FFNs
    float*  ybuf  = (float*) alloc((size_t)M_ * D_ * 4);          // GEMM2/4 out (pre-LN)
    int*    cum   = (int*)   alloc((size_t)B_ * L_ * 4);
    int*    idxb  = (int*)   alloc((size_t)B_ * MEL * 4);

    // --- length regulator (independent path) ---
    cumsum_kernel<<<1, 32, 0, stream>>>(target, cum);
    { int tot = B_ * MEL;           searchsorted_kernel<<<(tot + 255) / 256, 256, 0, stream>>>(cum, idxb); }
    { int tot = B_ * MEL * (D_/4);  gather_kernel<<<(tot + 255) / 256, 256, 0, stream>>>(x, idxb, cum, out); }

    // --- duration predictor ---
    { int tot = 3 * D_ * CD_; pack_w_kernel<D_, CD_><<<(tot + 255) / 256, 256, 0, stream>>>(w1, wf1); }
    { int tot = 3 * CD_ * D_; pack_w_kernel<CD_, D_><<<(tot + 255) / 256, 256, 0, stream>>>(w2, wf2); }
    { int tot = 3 * D_ * CD_; pack_w_kernel<D_, CD_><<<(tot + 255) / 256, 256, 0, stream>>>(w3, wf3); }
    { int tot = 3 * CD_ * D_; pack_w_kernel<CD_, D_><<<(tot + 255) / 256, 256, 0, stream>>>(w4, wf4); }
    { int tot = B_ * LPAD * D_; pad_x_kernel<<<(tot + 255) / 256, 256, 0, stream>>>(x, xpad); }
    { int tot = B_ * 2 * CD_;   zero_edges_kernel<<<(tot + 255) / 256, 256, 0, stream>>>(y1pad, CD_); }

    const int waves_big   = (M_ / 32) * (CD_ / 32);   // 12288 waves
    const int waves_small = (M_ / 32) * (D_  / 32);   // 3072 waves

    // FFN1: relu(conv 384->1536) -> relu(conv 1536->384)
    gemm_conv3_kernel<D_, CD_, true ><<<waves_big   * 32 / 256, 256, 0, stream>>>(xpad,  wf1, b1, y1pad, nullptr);
    gemm_conv3_kernel<CD_, D_, false><<<waves_small * 32 / 256, 256, 0, stream>>>(y1pad, wf2, b2, nullptr, ybuf);
    // LN1 -> bf16 padded (reuse xpad as h1pad); edges already zero but re-zero for safety
    { int tot = B_ * 2 * D_; zero_edges_kernel<<<(tot + 255) / 256, 256, 0, stream>>>(xpad, D_); }
    ln1_kernel<<<M_ * 32 / 256, 256, 0, stream>>>(ybuf, g1, be1, xpad);
    // FFN2
    gemm_conv3_kernel<D_, CD_, true ><<<waves_big   * 32 / 256, 256, 0, stream>>>(xpad,  wf3, b3, y1pad, nullptr);
    gemm_conv3_kernel<CD_, D_, false><<<waves_small * 32 / 256, 256, 0, stream>>>(y1pad, wf4, b4, nullptr, ybuf);
    // LN2 fused with linear head
    ln2_linear_kernel<<<M_ * 32 / 256, 256, 0, stream>>>(ybuf, g2, be2, wl, bl, dur);
}
